// GPT_523986010334
// MI455X (gfx1250) — matrix operand order repaired and checked by reading, not compiled, not run on hardware
//
#include <hip/hip_runtime.h>
#include <hip/hip_bf16.h>
#include <math.h>

#define TSEQ   1024
#define DMODEL 768
#define NHEAD  12
#define HDIM   64
#define QKVD   2304
#define FFDIM  3072
#define NLAYER 6
#define BBATCH 2
#define MROWS  (BBATCH * TSEQ)   // 2048
#define VOCAB  32000

typedef __attribute__((ext_vector_type(16))) __bf16       v16b;
typedef __attribute__((ext_vector_type(8)))  float        v8f;
typedef __attribute__((ext_vector_type(4)))  unsigned int v4u;
typedef __attribute__((ext_vector_type(4)))  float        v4f;

__device__ __forceinline__ __bf16 f2bf(float f) {
  unsigned int u = __float_as_uint(f);
  u = (u + 0x7FFFu + ((u >> 16) & 1u)) >> 16;   // round-to-nearest-even
  union { unsigned short s; __bf16 b; } c;
  c.s = (unsigned short)u;
  return c.b;
}

__device__ __forceinline__ unsigned int pack2(float a, float b) {
  union { __bf16 h[2]; unsigned int u; } p;
  p.h[0] = f2bf(a); p.h[1] = f2bf(b);
  return p.u;
}

// Build a 16-element bf16 fragment from two 16-byte LDS chunks (-> ds_load_b128 x2)
__device__ __forceinline__ v16b load_frag(const __bf16* p0, const __bf16* p1) {
  union { v4u q[2]; v16b v; } u;
  u.q[0] = *(const v4u*)p0;
  u.q[1] = *(const v4u*)p1;
  return u.v;
}

// Async 16-byte global -> LDS copy (per lane), tracked by ASYNCcnt.
__device__ __forceinline__ void async_ld16(__bf16* lds, const __bf16* gptr) {
  unsigned int loff = (unsigned int)(uintptr_t)lds;   // low 32 bits = LDS offset
  asm volatile("global_load_async_to_lds_b128 %0, %1, off"
               :: "v"(loff), "v"(gptr) : "memory");
}
__device__ __forceinline__ void async_wait0() {
  asm volatile("s_wait_asynccnt 0x0" ::: "memory");
}

// ---------------------------------------------------------------------------
// Embedding: x[b,t,:] = tok_emb[idx[b,t],:] + pos_emb[t,:]   (fp32 residual)
// ---------------------------------------------------------------------------
__global__ __launch_bounds__(256) void embed_kernel(
    const int* __restrict__ idx, const float* __restrict__ tok,
    const float* __restrict__ pos, float* __restrict__ x) {
  const int row = blockIdx.x;           // 0..2047
  const int t = row % TSEQ;
  const int id = idx[row];
  for (int j = threadIdx.x; j < DMODEL; j += 256)
    x[(size_t)row * DMODEL + j] = tok[(size_t)id * DMODEL + j] + pos[(size_t)t * DMODEL + j];
}

// ---------------------------------------------------------------------------
// LayerNorm fp32 in -> bf16 out: out = (x-mean)*rsqrt(var+1e-5)*w + b  (D=768)
// ---------------------------------------------------------------------------
__global__ __launch_bounds__(256) void ln_kernel(
    const float* __restrict__ x, const float* __restrict__ w,
    const float* __restrict__ b, __bf16* __restrict__ out) {
  const int row = blockIdx.x;
  const int tid = threadIdx.x;
  const float* xr = x + (size_t)row * DMODEL;
  float v0 = xr[tid], v1 = xr[tid + 256], v2 = xr[tid + 512];
  __shared__ float rs[256], rq[256];
  rs[tid] = v0 + v1 + v2;
  rq[tid] = v0 * v0 + v1 * v1 + v2 * v2;
  __syncthreads();
  for (int off = 128; off > 0; off >>= 1) {
    if (tid < off) { rs[tid] += rs[tid + off]; rq[tid] += rq[tid + off]; }
    __syncthreads();
  }
  const float mean = rs[0] * (1.0f / DMODEL);
  const float var  = rq[0] * (1.0f / DMODEL) - mean * mean;
  const float rstd = rsqrtf(var + 1e-5f);
  __bf16* orow = out + (size_t)row * DMODEL;
  orow[tid]       = f2bf((v0 - mean) * rstd * w[tid]       + b[tid]);
  orow[tid + 256] = f2bf((v1 - mean) * rstd * w[tid + 256] + b[tid + 256]);
  orow[tid + 512] = f2bf((v2 - mean) * rstd * w[tid + 512] + b[tid + 512]);
}

// ---------------------------------------------------------------------------
// Tiled GEMM: C[M,N] = A[M,K](bf16) @ W[N,K](fp32)^T, bf16 WMMA, fp32 accum.
// mode: bit0 +bias[n] | bit1 +R[m,n] fp32 residual | bit2 exact GELU | bit3 bf16 out
// Workgroup tile 128x128, 8 waves as 2(M)x4(N), each wave 4x2 16x16 tiles.
// A tile filled by async global->LDS b128 copies (ASYNCcnt), W tile by
// synchronous load+convert.
// ---------------------------------------------------------------------------
__global__ __launch_bounds__(256) void gemm_kernel(
    const __bf16* __restrict__ A, const float* __restrict__ W,
    const float* __restrict__ bias, const float* __restrict__ R,
    float* __restrict__ Cf, __bf16* __restrict__ Cb,
    int M, int N, int K, int mode) {
  __shared__ __align__(16) __bf16 As[128][40];   // 32 K + 8 pad (80B rows)
  __shared__ __align__(16) __bf16 Bs[128][40];
  const int tid  = threadIdx.x;
  const int wave = tid >> 5, lane = tid & 31;
  const int g = lane >> 4, nlo = lane & 15;
  const int waveM = wave >> 2, waveN = wave & 3;
  const int n0 = blockIdx.x * 128, m0 = blockIdx.y * 128;

  v8f zero = {};
  v8f acc[4][2];
#pragma unroll
  for (int i = 0; i < 4; ++i) { acc[i][0] = zero; acc[i][1] = zero; }

  const int trow = tid >> 1;
  const int tcol = (tid & 1) * 16;

  for (int k0 = 0; k0 < K; k0 += 32) {
    // A tile: 128 rows x 32 bf16 = 512 x 16B chunks, 2 per thread, async.
#pragma unroll
    for (int i = 0; i < 2; ++i) {
      const int c = tid + 256 * i;
      const int arow = c >> 2, sub = c & 3;
      async_ld16(&As[arow][sub * 8],
                 A + (size_t)(m0 + arow) * K + k0 + sub * 8);
    }
    // W tile: synchronous fp32 load + convert.
    const float* gw = W + (size_t)(n0 + trow) * K + (k0 + tcol);
#pragma unroll
    for (int j = 0; j < 16; j += 4) {
      v4f vw = *(const v4f*)(gw + j);
      *(unsigned int*)&Bs[trow][tcol + j]     = pack2(vw.x, vw.y);
      *(unsigned int*)&Bs[trow][tcol + j + 2] = pack2(vw.z, vw.w);
    }
    async_wait0();
    __syncthreads();

    v16b bfrag[2];
#pragma unroll
    for (int nt = 0; nt < 2; ++nt) {
      const __bf16* bp = &Bs[waveN * 32 + nt * 16 + nlo][g * 8];
      bfrag[nt] = load_frag(bp, bp + 16);
    }
#pragma unroll
    for (int mt = 0; mt < 4; ++mt) {
      const __bf16* ap = &As[waveM * 64 + mt * 16 + nlo][0];
      v16b afrag = load_frag(ap + g * 8, ap + 16 + g * 8);
#pragma unroll
      for (int nt = 0; nt < 2; ++nt) {
        acc[mt][nt] = __builtin_amdgcn_wmma_f32_16x16x32_bf16(
            false, afrag, false, bfrag[nt], (short)0, acc[mt][nt], false, false);
      }
    }
    __syncthreads();
  }

#pragma unroll
  for (int mt = 0; mt < 4; ++mt) {
#pragma unroll
    for (int nt = 0; nt < 2; ++nt) {
#pragma unroll
      for (int r = 0; r < 8; ++r) {
        const int m = m0 + waveM * 64 + mt * 16 + r + 8 * g;
        const int n = n0 + waveN * 32 + nt * 16 + nlo;
        float v = acc[mt][nt][r];
        if (mode & 1) v += bias[n];
        if (mode & 4) v = 0.5f * v * (1.0f + erff(v * 0.70710678118654752f));
        if (mode & 2) v += R[(size_t)m * N + n];
        if (mode & 8) Cb[(size_t)m * N + n] = f2bf(v);
        else          Cf[(size_t)m * N + n] = v;
      }
    }
  }
}

// ---------------------------------------------------------------------------
// Causal flash attention, bf16 qkv in / bf16 y out.
// qkv layout per reference: [B,T, h*192 + {q:0,k:64,v:128}+d]
// One block per (query-tile of 128, b*H+h). Wave w owns q rows w*16..w*16+15.
// Key tiles of 64, online softmax, O accumulated in fp32.
// ---------------------------------------------------------------------------
__global__ __launch_bounds__(256) void attn_kernel(
    const __bf16* __restrict__ qkv, __bf16* __restrict__ y) {
  __shared__ __align__(16) __bf16 Qs[128][72];    // [q][d]
  __shared__ __align__(16) __bf16 Ks[64][72];     // [k][d]
  __shared__ __align__(16) __bf16 Vt[64][72];     // [d][k]  (transposed)
  __shared__ __align__(16) __bf16 Ps[8][16][72];  // per-wave P [q][k]
  const int tid  = threadIdx.x;
  const int wave = tid >> 5, lane = tid & 31;
  const int g = lane >> 4, nlo = lane & 15;
  const int qt = blockIdx.x;
  const int bb = blockIdx.y / NHEAD, hh = blockIdx.y % NHEAD;
  const size_t rowbase = (size_t)bb * TSEQ;

  // Q tile 128x64 bf16: 1024 x 16B chunks, 4 per thread, async.
#pragma unroll
  for (int i = 0; i < 4; ++i) {
    const int c = tid + 256 * i;
    const int qrow = c >> 3, sub = c & 7;
    async_ld16(&Qs[qrow][sub * 8],
               qkv + (rowbase + qt * 128 + qrow) * QKVD + hh * 192 + sub * 8);
  }
  async_wait0();
  __syncthreads();

  const int qrow = wave * 16 + nlo;
  v16b aq[2];
#pragma unroll
  for (int ks = 0; ks < 2; ++ks) {
    const __bf16* p = &Qs[qrow][ks * 32 + g * 8];
    aq[ks] = load_frag(p, p + 16);
  }

  v8f zero = {};
  v8f o[4]; o[0] = zero; o[1] = zero; o[2] = zero; o[3] = zero;
  float mrow[8], lrow[8];
#pragma unroll
  for (int r = 0; r < 8; ++r) { mrow[r] = -1e30f; lrow[r] = 0.0f; }

  const int ktiles = 2 * (qt + 1);
  for (int kt = 0; kt < ktiles; ++kt) {
    const int kt0 = kt * 64;
    __syncthreads();   // previous tile fully consumed
    // K tile 64x64: 512 x 16B chunks, 2 per thread, async.
#pragma unroll
    for (int i = 0; i < 2; ++i) {
      const int c = tid + 256 * i;
      const int krow = c >> 3, sub = c & 7;
      async_ld16(&Ks[krow][sub * 8],
                 qkv + (rowbase + kt0 + krow) * QKVD + hh * 192 + 64 + sub * 8);
    }
    { // V tile transposed: strided bf16 copies
      const int d = tid >> 2;
      const int cb = (tid & 3) * 16;
#pragma unroll
      for (int j = 0; j < 16; ++j)
        Vt[d][cb + j] = qkv[(rowbase + kt0 + cb + j) * QKVD + hh * 192 + 128 + d];
    }
    async_wait0();
    __syncthreads();

    // S = Q K^T (16x64 per wave)
    v8f s[4]; s[0] = zero; s[1] = zero; s[2] = zero; s[3] = zero;
#pragma unroll
    for (int nt = 0; nt < 4; ++nt) {
#pragma unroll
      for (int ks = 0; ks < 2; ++ks) {
        const __bf16* bp = &Ks[nt * 16 + nlo][ks * 32 + g * 8];
        v16b bk = load_frag(bp, bp + 16);
        s[nt] = __builtin_amdgcn_wmma_f32_16x16x32_bf16(
            false, aq[ks], false, bk, (short)0, s[nt], false, false);
      }
    }
    // scale + causal mask
    const int tqb = qt * 128 + wave * 16 + 8 * g;  // + r
#pragma unroll
    for (int nt = 0; nt < 4; ++nt) {
      const int tka = kt0 + nt * 16 + nlo;
#pragma unroll
      for (int r = 0; r < 8; ++r) {
        float v = s[nt][r] * 0.125f;
        if (tka > tqb + r) v = -1e30f;
        s[nt][r] = v;
      }
    }
    // online softmax (rows live in 16-lane halves)
#pragma unroll
    for (int r = 0; r < 8; ++r) {
      float mx = fmaxf(fmaxf(s[0][r], s[1][r]), fmaxf(s[2][r], s[3][r]));
      mx = fmaxf(mx, __shfl_xor(mx, 8));
      mx = fmaxf(mx, __shfl_xor(mx, 4));
      mx = fmaxf(mx, __shfl_xor(mx, 2));
      mx = fmaxf(mx, __shfl_xor(mx, 1));
      const float mnew  = fmaxf(mrow[r], mx);
      const float alpha = __expf(mrow[r] - mnew);
      float sum = 0.0f;
#pragma unroll
      for (int nt = 0; nt < 4; ++nt) {
        float p = __expf(s[nt][r] - mnew);
        s[nt][r] = p;
        sum += p;
      }
      sum += __shfl_xor(sum, 8);
      sum += __shfl_xor(sum, 4);
      sum += __shfl_xor(sum, 2);
      sum += __shfl_xor(sum, 1);
      lrow[r] = lrow[r] * alpha + sum;
      mrow[r] = mnew;
      o[0][r] *= alpha; o[1][r] *= alpha; o[2][r] *= alpha; o[3][r] *= alpha;
    }
    // stage P (bf16) in per-wave LDS (wave-private: in-order DS, no barrier)
#pragma unroll
    for (int nt = 0; nt < 4; ++nt) {
#pragma unroll
      for (int r = 0; r < 8; ++r)
        Ps[wave][r + 8 * g][nt * 16 + nlo] = f2bf(s[nt][r]);
    }
    // O += P @ V
#pragma unroll
    for (int no = 0; no < 4; ++no) {
#pragma unroll
      for (int ks = 0; ks < 2; ++ks) {
        const __bf16* ap = &Ps[wave][nlo][ks * 32 + g * 8];
        v16b af = load_frag(ap, ap + 16);
        const __bf16* vp = &Vt[no * 16 + nlo][ks * 32 + g * 8];
        v16b bv = load_frag(vp, vp + 16);
        o[no] = __builtin_amdgcn_wmma_f32_16x16x32_bf16(
            false, af, false, bv, (short)0, o[no], false, false);
      }
    }
  }
  // y[b, tq, hh*64 + d] = O / l
#pragma unroll
  for (int no = 0; no < 4; ++no) {
#pragma unroll
    for (int r = 0; r < 8; ++r) {
      const int tq = qt * 128 + wave * 16 + r + 8 * g;
      const int d  = hh * HDIM + no * 16 + nlo;
      y[(rowbase + tq) * DMODEL + d] = f2bf(o[no][r] / lrow[r]);
    }
  }
}

// ---------------------------------------------------------------------------
extern "C" void kernel_launch(void* const* d_in, const int* in_sizes, int n_in,
                              void* d_out, int out_size, void* d_ws, size_t ws_size,
                              hipStream_t stream) {
  (void)in_sizes; (void)n_in; (void)out_size; (void)ws_size;
  const int*   idx     = (const int*)  d_in[0];
  const float* tok_emb = (const float*)d_in[1];
  const float* pos_emb = (const float*)d_in[2];
  const float* ln1_w   = (const float*)d_in[3];
  const float* ln1_b   = (const float*)d_in[4];
  const float* qkv_w   = (const float*)d_in[5];
  const float* qkv_b   = (const float*)d_in[6];
  const float* out_w   = (const float*)d_in[7];
  const float* out_b   = (const float*)d_in[8];
  const float* ln2_w   = (const float*)d_in[9];
  const float* ln2_b   = (const float*)d_in[10];
  const float* fc1_w   = (const float*)d_in[11];
  const float* fc1_b   = (const float*)d_in[12];
  const float* fc2_w   = (const float*)d_in[13];
  const float* fc2_b   = (const float*)d_in[14];
  const float* lnf_w   = (const float*)d_in[15];
  const float* lnf_b   = (const float*)d_in[16];
  const float* head_w  = (const float*)d_in[17];
  float* logits = (float*)d_out;

  // Workspace layout: fp32 residual stream + bf16 activations.
  float*  x     = (float*)d_ws;                       // [2048,768] fp32
  __bf16* hbuf  = (__bf16*)(x + (size_t)MROWS * DMODEL);       // [2048,768]
  __bf16* qkvb  = hbuf + (size_t)MROWS * DMODEL;               // [2048,2304]
  __bf16* ybuf  = qkvb + (size_t)MROWS * QKVD;                 // [2048,768]
  __bf16* ffb   = ybuf + (size_t)MROWS * DMODEL;               // [2048,3072]

  dim3 blk(256);
  embed_kernel<<<MROWS, blk, 0, stream>>>(idx, tok_emb, pos_emb, x);

  for (int l = 0; l < NLAYER; ++l) {
    ln_kernel<<<MROWS, blk, 0, stream>>>(x, ln1_w + (size_t)l * DMODEL,
                                         ln1_b + (size_t)l * DMODEL, hbuf);
    gemm_kernel<<<dim3(QKVD / 128, MROWS / 128), blk, 0, stream>>>(
        hbuf, qkv_w + (size_t)l * QKVD * DMODEL, qkv_b + (size_t)l * QKVD,
        nullptr, nullptr, qkvb, MROWS, QKVD, DMODEL, /*bias|bf16out*/ 1 | 8);
    attn_kernel<<<dim3(TSEQ / 128, BBATCH * NHEAD), blk, 0, stream>>>(qkvb, ybuf);
    gemm_kernel<<<dim3(DMODEL / 128, MROWS / 128), blk, 0, stream>>>(
        ybuf, out_w + (size_t)l * DMODEL * DMODEL, out_b + (size_t)l * DMODEL,
        x, x, nullptr, MROWS, DMODEL, DMODEL, /*bias|res*/ 3);
    ln_kernel<<<MROWS, blk, 0, stream>>>(x, ln2_w + (size_t)l * DMODEL,
                                         ln2_b + (size_t)l * DMODEL, hbuf);
    gemm_kernel<<<dim3(FFDIM / 128, MROWS / 128), blk, 0, stream>>>(
        hbuf, fc1_w + (size_t)l * FFDIM * DMODEL, fc1_b + (size_t)l * FFDIM,
        nullptr, nullptr, ffb, MROWS, FFDIM, DMODEL, /*bias|gelu|bf16out*/ 1 | 4 | 8);
    gemm_kernel<<<dim3(DMODEL / 128, MROWS / 128), blk, 0, stream>>>(
        ffb, fc2_w + (size_t)l * DMODEL * FFDIM, fc2_b + (size_t)l * DMODEL,
        x, x, nullptr, MROWS, DMODEL, FFDIM, /*bias|res*/ 3);
  }

  ln_kernel<<<MROWS, blk, 0, stream>>>(x, lnf_w, lnf_b, hbuf);
  gemm_kernel<<<dim3(VOCAB / 128, MROWS / 128), blk, 0, stream>>>(
      hbuf, head_w, nullptr, nullptr, logits, nullptr,
      MROWS, VOCAB, DMODEL, /*plain fp32 out*/ 0);
}
